// CompiledNCA_61297773248858
// MI455X (gfx1250) — matrix operand order repaired and checked by reading, not compiled, not run on hardware
//
#include <hip/hip_runtime.h>

// ---------------------------------------------------------------------------
// NCA on MI455X (gfx1250): 16 iterated 3x3 conv (16->16 ch) as implicit GEMM
// via v_wmma_f32_16x16x32_bf16. h stored NHWC bf16 (32B/pixel) to halve HBM
// traffic (memory-bound workload: ~155 GFLOP vs ~2.2GB -> ~95us at 23.3TB/s).
// Activation tiles staged into LDS with global_load_async_to_lds_b128.
// ---------------------------------------------------------------------------

typedef __attribute__((ext_vector_type(16))) __bf16     v16bf;
typedef __attribute__((ext_vector_type(8)))  float      v8f;
typedef __attribute__((ext_vector_type(4)))  unsigned   v4u;

// Types for the async-to-LDS builtin (signature revealed by clang diagnostic:
// param0 = int4 in device/global AS).
typedef int v4i_vs __attribute__((vector_size(16)));
typedef __attribute__((address_space(1))) v4i_vs* gas_ptr;
typedef __attribute__((address_space(3))) v4i_vs* las_ptr;

#define H_IMG 128
#define W_IMG 128
#define B_IMG 128
#define HC    16
#define T_STEPS 16
#define TILE  32
#define HALO  34            // TILE + 2
#define PIX_BYTES 32        // 16 ch * 2B bf16
#define IMG_PIX (H_IMG * W_IMG)
#define H_BYTES ((size_t)B_IMG * IMG_PIX * PIX_BYTES)   // 64 MB per buffer
#define WPK_DW_PER_STEP 1280   // 5 tap-pairs * 32 lanes * 8 dwords

// ---- async global->LDS (guarded so compile can't break on absence) --------
#if defined(__has_builtin)
#if __has_builtin(__builtin_amdgcn_global_load_async_to_lds_b128) && \
    __has_builtin(__builtin_amdgcn_s_wait_asynccnt)
#define NCA_ASYNC_LDS 1
#endif
#endif
#ifndef NCA_ASYNC_LDS
#define NCA_ASYNC_LDS 0
#endif

__device__ __forceinline__ unsigned short f2bf(float f) {
  unsigned u = __float_as_uint(f);
  u += 0x7FFFu + ((u >> 16) & 1u);   // round-to-nearest-even
  return (unsigned short)(u >> 16);
}

// ---------------------------------------------------------------------------
// Pack weight_schedule (T,16,16,3,3) f32 into per-lane bf16 B-fragments.
// B layout (32x16, K x N, 8 VGPRs): lane n<16 -> col n, K=0..7,16..23 (2/VGPR);
// lane n+16 -> col n, K=8..15,24..31. K 0..15 = in-ch of tap0, K16..31 = tap1.
// Tap-pairs: tp0..3 = taps (2tp,2tp+1); tp4 = tap8 + zero pad.
// Output: wpk[step][tp][lane][8 dwords]
// ---------------------------------------------------------------------------
__global__ __launch_bounds__(256) void nca_pack_w(
    const float* __restrict__ wsched, unsigned* __restrict__ wpk) {
  const int s = blockIdx.x;
  for (int e = threadIdx.x; e < WPK_DW_PER_STEP; e += 256) {
    const int v    = e & 7;
    const int lane = (e >> 3) & 31;
    const int tp   = e >> 8;
    const int col  = lane & 15;    // out-channel
    const int half = lane >> 4;    // which in-channel half (0..7 / 8..15)
    int tap; bool valid = true;
    if (tp < 4) tap = 2 * tp + (v >> 2);
    else { if (v < 4) tap = 8; else { tap = 0; valid = false; } }
    unsigned val = 0u;
    if (valid) {
      const int c0 = 8 * half + 2 * (v & 3), c1 = c0 + 1;
      const float f0 = wsched[((s * HC + col) * HC + c0) * 9 + tap];
      const float f1 = wsched[((s * HC + col) * HC + c1) * 9 + tap];
      val = (unsigned)f2bf(f0) | ((unsigned)f2bf(f1) << 16);
    }
    wpk[s * WPK_DW_PER_STEP + e] = val;
  }
}

// ---------------------------------------------------------------------------
// Stem: 3x3 conv 1->16 + bias + relu, f32 in -> bf16 NHWC out. VALU-bound but
// tiny (<0.5% of total FLOPs).
// ---------------------------------------------------------------------------
__global__ __launch_bounds__(256) void nca_stem(
    const float* __restrict__ x, const float* __restrict__ w,
    const float* __restrict__ bias, unsigned char* __restrict__ h0) {
  __shared__ float sw[160];
  const int tid = threadIdx.x;
  if (tid < 144)      sw[tid] = w[tid];
  else if (tid < 160) sw[tid] = bias[tid - 144];
  __syncthreads();

  const int p   = blockIdx.x * 256 + tid;     // global pixel
  const int img = p >> 14;
  const int rem = p & (IMG_PIX - 1);
  const int y   = rem >> 7;
  const int xx  = rem & 127;

  float in[9];
#pragma unroll
  for (int t = 0; t < 9; ++t) {
    const int gy = y + t / 3 - 1, gx = xx + t % 3 - 1;
    const bool ok = (unsigned)gy < (unsigned)H_IMG && (unsigned)gx < (unsigned)W_IMG;
    in[t] = ok ? x[img * IMG_PIX + gy * W_IMG + gx] : 0.f;
  }

  v4u pk0, pk1;
#pragma unroll
  for (int cp = 0; cp < 8; ++cp) {
    unsigned v = 0u;
#pragma unroll
    for (int hh = 0; hh < 2; ++hh) {
      const int c = cp * 2 + hh;
      float acc = sw[144 + c];
#pragma unroll
      for (int t = 0; t < 9; ++t) acc += sw[c * 9 + t] * in[t];
      acc = acc > 0.f ? acc : 0.f;
      v |= ((unsigned)f2bf(acc)) << (hh * 16);
    }
    if (cp < 4) pk0[cp] = v; else pk1[cp - 4] = v;
  }
  v4u* dst = (v4u*)(h0 + (size_t)p * PIX_BYTES);
  dst[0] = pk0;
  dst[1] = pk1;
}

// ---------------------------------------------------------------------------
// One conv step: each block handles a 32x32 pixel tile of one image.
// 8 waves x 8 pixel-groups, 5 bf16 WMMAs per 16-pixel group (K=144 pad 160).
// ---------------------------------------------------------------------------
__global__ __launch_bounds__(256) void nca_conv_step(
    const unsigned char* __restrict__ hin,
    unsigned char* __restrict__ hout,
    const unsigned* __restrict__ wstep) {
  __shared__ __align__(16) unsigned char tile[HALO * HALO * PIX_BYTES];

  const int tid = threadIdx.x;
  const int bx  = blockIdx.x;
  const int img = bx >> 4;
  const int tt  = bx & 15;
  const int y0  = (tt >> 2) * TILE;
  const int x0  = (tt & 3) * TILE;
  const int lane = tid & 31;
  const v4u zero4 = {0u, 0u, 0u, 0u};

  // Preload the 5 B (weight) fragments: 32B/lane each, packed layout.
  union BF { v16bf bf; v4u u[2]; };
  BF bw[5];
#pragma unroll
  for (int tp = 0; tp < 5; ++tp) {
    const v4u* wp = (const v4u*)(wstep + (tp * 32 + lane) * 8);
    bw[tp].u[0] = wp[0];
    bw[tp].u[1] = wp[1];
  }

  // Stage haloed input tile (NHWC bf16) into LDS, zero-filling OOB pixels.
  const unsigned char* imgbase = hin + (size_t)img * IMG_PIX * PIX_BYTES;
#if NCA_ASYNC_LDS
  for (int idx = tid; idx < HALO * HALO; idx += 256) {
    const int ty = idx / HALO, tx = idx % HALO;
    const int gy = y0 + ty - 1, gx = x0 + tx - 1;
    if ((unsigned)gy < (unsigned)H_IMG && (unsigned)gx < (unsigned)W_IMG) {
      const unsigned char* g = imgbase + (size_t)(gy * W_IMG + gx) * PIX_BYTES;
      __builtin_amdgcn_global_load_async_to_lds_b128(
          (gas_ptr)g, (las_ptr)(tile + idx * PIX_BYTES), 0, 0);
      __builtin_amdgcn_global_load_async_to_lds_b128(
          (gas_ptr)(g + 16), (las_ptr)(tile + idx * PIX_BYTES + 16), 0, 0);
    } else {
      v4u* dst = (v4u*)(tile + idx * PIX_BYTES);
      dst[0] = zero4; dst[1] = zero4;
    }
  }
  __builtin_amdgcn_s_wait_asynccnt(0);
  __syncthreads();
#else
  for (int idx = tid; idx < HALO * HALO; idx += 256) {
    const int ty = idx / HALO, tx = idx % HALO;
    const int gy = y0 + ty - 1, gx = x0 + tx - 1;
    v4u a0 = zero4, a1 = zero4;
    if ((unsigned)gy < (unsigned)H_IMG && (unsigned)gx < (unsigned)W_IMG) {
      const v4u* src = (const v4u*)(imgbase + (size_t)(gy * W_IMG + gx) * PIX_BYTES);
      a0 = src[0]; a1 = src[1];
    }
    v4u* dst = (v4u*)(tile + idx * PIX_BYTES);
    dst[0] = a0; dst[1] = a1;
  }
  __syncthreads();
#endif

  const int wv    = tid >> 5;
  const int l     = lane & 15;   // pixel within group for A; out-channel for D
  const int lhalf = lane >> 4;
  const int chb   = lhalf * 16;  // A-fragment channel byte offset (ch 0..7 / 8..15)

#pragma unroll
  for (int rr = 0; rr < 4; ++rr) {
    const int r = wv * 4 + rr;            // row within tile (0..31)
#pragma unroll
    for (int cg = 0; cg < 2; ++cg) {
      const int c0 = cg * 16;             // col-group base
      v8f acc = {0.f, 0.f, 0.f, 0.f, 0.f, 0.f, 0.f, 0.f};
#pragma unroll
      for (int tp = 0; tp < 5; ++tp) {
        union AF { v16bf bf; v4u u[2]; } a;
        const int t0 = 2 * tp;
        const int off0 = ((r + t0 / 3) * HALO + (c0 + l + t0 % 3)) * PIX_BYTES + chb;
        a.u[0] = *(const v4u*)(tile + off0);
        if (tp < 4) {
          const int t1 = t0 + 1;
          const int off1 = ((r + t1 / 3) * HALO + (c0 + l + t1 % 3)) * PIX_BYTES + chb;
          a.u[1] = *(const v4u*)(tile + off1);
        } else {
          a.u[1] = zero4;
        }
        acc = __builtin_amdgcn_wmma_f32_16x16x32_bf16(
            false, a.bf, false, bw[tp].bf, (short)0, acc, false, false);
      }
      // ReLU + bf16 store. D layout: VGPR rv -> pixel rv (lanes 0-15) /
      // rv+8 (lanes 16-31); channel = lane&15. Stores coalesce to 32B runs.
      const int ybase = y0 + r;
      unsigned short* outp = (unsigned short*)hout;
#pragma unroll
      for (int rv = 0; rv < 8; ++rv) {
        const int m = rv + lhalf * 8;
        float v = acc[rv];
        v = v > 0.f ? v : 0.f;
        const int off = ((img * H_IMG + ybase) * W_IMG + (x0 + c0 + m)) * HC + l;
        outp[off] = f2bf(v);
      }
    }
  }
}

// ---------------------------------------------------------------------------
// Mean over 128x128 spatial + FC (16->10). One block per image.
// ---------------------------------------------------------------------------
__global__ __launch_bounds__(256) void nca_pool_fc(
    const unsigned char* __restrict__ h, const float* __restrict__ fcw,
    const float* __restrict__ fcb, float* __restrict__ out) {
  __shared__ float sm[256 * HC];
  __shared__ float pooled[HC];
  const int tid = threadIdx.x;
  const int b   = blockIdx.x;

  float acc[HC];
#pragma unroll
  for (int c = 0; c < HC; ++c) acc[c] = 0.f;

  const v4u* base = (const v4u*)(h + (size_t)b * IMG_PIX * PIX_BYTES);
  for (int p = tid; p < IMG_PIX; p += 256) {
    const v4u u0 = base[p * 2];
    const v4u u1 = base[p * 2 + 1];
#pragma unroll
    for (int i = 0; i < 4; ++i) {
      const unsigned a = u0[i], bb = u1[i];
      acc[2 * i]         += __uint_as_float(a << 16);
      acc[2 * i + 1]     += __uint_as_float(a & 0xffff0000u);
      acc[8 + 2 * i]     += __uint_as_float(bb << 16);
      acc[8 + 2 * i + 1] += __uint_as_float(bb & 0xffff0000u);
    }
  }
#pragma unroll
  for (int c = 0; c < HC; ++c) sm[tid * HC + c] = acc[c];
  __syncthreads();

  if (tid < HC) {
    float s = 0.f;
    for (int i = 0; i < 256; ++i) s += sm[i * HC + tid];
    pooled[tid] = s * (1.0f / (float)IMG_PIX);
  }
  __syncthreads();

  if (tid < 10) {
    float o = fcb[tid];
#pragma unroll
    for (int c = 0; c < HC; ++c) o += pooled[c] * fcw[tid * HC + c];
    out[b * 10 + tid] = o;
  }
}

// ---------------------------------------------------------------------------
extern "C" void kernel_launch(void* const* d_in, const int* in_sizes, int n_in,
                              void* d_out, int out_size, void* d_ws, size_t ws_size,
                              hipStream_t stream) {
  const float* x      = (const float*)d_in[0];
  const float* stem_w = (const float*)d_in[1];
  const float* stem_b = (const float*)d_in[2];
  const float* wsched = (const float*)d_in[3];
  const float* fcw    = (const float*)d_in[4];
  const float* fcb    = (const float*)d_in[5];

  unsigned char* ws = (unsigned char*)d_ws;
  unsigned char* h0 = ws;
  unsigned char* h1 = ws + H_BYTES;
  unsigned* wpk = (unsigned*)(ws + 2 * H_BYTES);

  nca_pack_w<<<T_STEPS, 256, 0, stream>>>(wsched, wpk);
  nca_stem<<<(B_IMG * IMG_PIX) / 256, 256, 0, stream>>>(x, stem_w, stem_b, h0);

  for (int s = 0; s < T_STEPS; ++s) {
    const unsigned char* hin = (s & 1) ? h1 : h0;
    unsigned char* hout      = (s & 1) ? h0 : h1;
    nca_conv_step<<<B_IMG * 16, 256, 0, stream>>>(hin, hout,
                                                  wpk + s * WPK_DW_PER_STEP);
  }
  // T even -> final state in h0
  nca_pool_fc<<<B_IMG, 256, 0, stream>>>(h0, fcw, fcb, (float*)d_out);
}
